// SimpleLSTMDecoder_16372415332771
// MI455X (gfx1250) — matrix operand order repaired and checked by reading, not compiled, not run on hardware
//
#include <hip/hip_runtime.h>
#include <hip/hip_bf16.h>
#include <math.h>

// Problem dims (match reference)
#define B_    32
#define T_    64
#define S_    256
#define DIN_  1024
#define H_    1024
#define V_    20000
#define OOV_  50
#define VT_   (V_ + OOV_)

typedef __bf16 bf16_t;
typedef __attribute__((ext_vector_type(16))) __bf16 v16bf;
typedef __attribute__((ext_vector_type(8)))  __bf16 v8bf;
typedef __attribute__((ext_vector_type(8)))  float  v8f;

// ---------------------------------------------------------------------------
// f32 -> bf16 convert
// ---------------------------------------------------------------------------
__global__ void cvt_bf16_kernel(const float* __restrict__ in,
                                bf16_t* __restrict__ out, int n) {
  int i = blockIdx.x * blockDim.x + threadIdx.x;
  if (i < n) out[i] = (bf16_t)in[i];
}

__global__ void add_bias_kernel(const float* __restrict__ a,
                                const float* __restrict__ b,
                                float* __restrict__ o, int n) {
  int i = blockIdx.x * blockDim.x + threadIdx.x;
  if (i < n) o[i] = a[i] + b[i];
}

// ---------------------------------------------------------------------------
// WMMA GEMM:  C[M,N] = A[M,K] @ W[N,K]^T (+bias[N]) (+=C if ACCUM)
// A, W bf16 row-major; C f32 row-major.  M multiple of 16*MT, N of 16, K of 32.
// One wave computes MT vertically-stacked 16x16 output tiles, reusing each
// loaded B (weight) fragment MT times -> halves L2 weight traffic for MT=2.
// 4 waves per block tile N.
//
// VGPR layouts used here (wave32; A/B in contiguous k order, not the ISA table order):
//   A (16x32 bf16): lane L -> row m=L%16, kh=L/16; elems 0..7 = K=kh*16+0..7,
//                   elems 8..15 = K=kh*16+8+0..7  (two 16B loads)
//   B (32x16 bf16): lane L -> col n=L%16, kh=L/16; elems 0..15 = K=kh*16+0..15
//                   (one 32B load; W row n is contiguous K)
//   C/D (16x16 f32): vgpr r, lanes 0-15 -> (m=r, n=lane),
//                    lanes 16-31 -> (m=r+8, n=lane-16)
// ---------------------------------------------------------------------------
template <bool HAS_BIAS, bool ACCUM, int MT>
__global__ void wmma_gemm_kernel(const bf16_t* __restrict__ A,
                                 const bf16_t* __restrict__ W,
                                 const float* __restrict__ bias,
                                 float* __restrict__ C,
                                 int M, int N, int K) {
  const int lane = threadIdx.x & 31;
  const int wave = threadIdx.x >> 5;
  const int ln   = lane & 15;
  const int kh   = lane >> 4;   // 0 or 1
  const int tn   = (blockIdx.x * (blockDim.x >> 5) + wave) * 16;
  const int tm   = blockIdx.y * (16 * MT);
  if (tn >= N) return;

  v8f acc[MT];
#pragma unroll
  for (int mt = 0; mt < MT; ++mt) {
#pragma unroll
    for (int r = 0; r < 8; ++r) {
      if (ACCUM) {
        int row = tm + mt * 16 + r + (kh ? 8 : 0);
        acc[mt][r] = C[(size_t)row * N + tn + ln];
      } else {
        acc[mt][r] = 0.0f;
      }
    }
  }

  const bf16_t* pa = A + (size_t)(tm + ln) * K + kh * 16;
  const bf16_t* pw = W + (size_t)(tn + ln) * K + kh * 16;

  for (int k0 = 0; k0 < K; k0 += 32) {
    v16bf bv = *(const v16bf*)(pw + k0);
#pragma unroll
    for (int mt = 0; mt < MT; ++mt) {
      v8bf alo = *(const v8bf*)(pa + (size_t)mt * 16 * K + k0);
      v8bf ahi = *(const v8bf*)(pa + (size_t)mt * 16 * K + k0 + 8);
      v16bf av;
#pragma unroll
      for (int i = 0; i < 8; ++i) { av[i] = alo[i]; av[i + 8] = ahi[i]; }
      acc[mt] = __builtin_amdgcn_wmma_f32_16x16x32_bf16(
          false, av, false, bv, (short)0, acc[mt], false, false);
    }
  }

#pragma unroll
  for (int mt = 0; mt < MT; ++mt) {
#pragma unroll
    for (int r = 0; r < 8; ++r) {
      int row   = tm + mt * 16 + r + (kh ? 8 : 0);
      float val = acc[mt][r];
      if (HAS_BIAS) val += bias[tn + ln];
      C[(size_t)row * N + tn + ln] = val;
    }
  }
}

// ---------------------------------------------------------------------------
// Attention score: sc[b,s] = mask ? sum_h tanh(hq[b,h]+keys[b,s,h])*v[h] : -1e9
// One block per (b,s), 256 threads.
// ---------------------------------------------------------------------------
__global__ void attn_score_kernel(const float* __restrict__ hq,
                                  const float* __restrict__ keys,
                                  const float* __restrict__ v,
                                  const unsigned char* __restrict__ mask,
                                  float* __restrict__ sc) {
  int bs = blockIdx.x;
  int b  = bs / S_;
  const float* kq = keys + (size_t)bs * H_;
  const float* hb = hq + (size_t)b * H_;
  float sum = 0.0f;
  for (int h = threadIdx.x; h < H_; h += 256)
    sum += tanhf(hb[h] + kq[h]) * v[h];
  __shared__ float red[256];
  red[threadIdx.x] = sum;
  __syncthreads();
  for (int off = 128; off > 0; off >>= 1) {
    if (threadIdx.x < off) red[threadIdx.x] += red[threadIdx.x + off];
    __syncthreads();
  }
  if (threadIdx.x == 0) sc[bs] = mask[bs] ? red[0] : -1e9f;
}

// softmax over S (=256) per row; in place. blockDim.x == 256.
__global__ void softmax_s_kernel(float* __restrict__ sc) {
  int b = blockIdx.x;
  float* p = sc + (size_t)b * S_;
  int t = threadIdx.x;
  float x = p[t];
  __shared__ float red[256];
  red[t] = x;
  __syncthreads();
  for (int off = 128; off > 0; off >>= 1) {
    if (t < off) red[t] = fmaxf(red[t], red[t + off]);
    __syncthreads();
  }
  float m = red[0];
  __syncthreads();
  float e = __expf(x - m);
  red[t] = e;
  __syncthreads();
  for (int off = 128; off > 0; off >>= 1) {
    if (t < off) red[t] += red[t + off];
    __syncthreads();
  }
  p[t] = e / red[0];
}

// ctx[b,h] = sum_s alpha[b,s] * nodes[b,s,h].  One block per b, 256 threads.
__global__ void ctx_kernel(const float* __restrict__ alpha,
                           const float* __restrict__ nodes,
                           float* __restrict__ ctx) {
  int b = blockIdx.x;
  __shared__ float a[S_];
  for (int s = threadIdx.x; s < S_; s += blockDim.x) a[s] = alpha[b * S_ + s];
  __syncthreads();
  for (int h = threadIdx.x; h < H_; h += blockDim.x) {
    float sum = 0.0f;
    const float* np = nodes + (size_t)b * S_ * H_ + h;
    for (int s = 0; s < S_; ++s) sum += a[s] * np[(size_t)s * H_];
    ctx[b * H_ + h] = sum;
  }
}

// xin_bf[b, 0:DIN]=x_t[b], [DIN:DIN+H]=ctx[b], converted to bf16.
__global__ void concat_kernel(const float* __restrict__ x,
                              const float* __restrict__ ctx,
                              bf16_t* __restrict__ xin, int t) {
  int i = blockIdx.x * blockDim.x + threadIdx.x;  // B*(DIN+H)
  if (i >= B_ * (DIN_ + H_)) return;
  int b = i / (DIN_ + H_);
  int j = i % (DIN_ + H_);
  float v = (j < DIN_) ? x[((size_t)b * T_ + t) * DIN_ + j]
                       : ctx[b * H_ + (j - DIN_)];
  xin[i] = (bf16_t)v;
}

// LSTM cell (PyTorch gate order i,f,g,o). Writes h (f32), c (f32), h_bf.
__global__ void lstm_cell_kernel(const float* __restrict__ gates,
                                 float* __restrict__ h, float* __restrict__ c,
                                 bf16_t* __restrict__ h_bf) {
  int i = blockIdx.x * blockDim.x + threadIdx.x;  // B*H
  if (i >= B_ * H_) return;
  int b = i / H_, j = i % H_;
  const float* g = gates + (size_t)b * 4 * H_;
  float ig = 1.0f / (1.0f + __expf(-g[j]));
  float fg = 1.0f / (1.0f + __expf(-g[H_ + j]));
  float gg = tanhf(g[2 * H_ + j]);
  float og = 1.0f / (1.0f + __expf(-g[3 * H_ + j]));
  float cn = fg * c[i] + ig * gg;
  float hn = og * tanhf(cn);
  c[i] = cn;
  h[i] = hn;
  h_bf[i] = (bf16_t)hn;
}

// p_gen[b] = sigmoid(h[b,:] . W_sw + b_sw). One block per b.
__global__ void pgen_kernel(const float* __restrict__ h,
                            const float* __restrict__ W_sw,
                            const float* __restrict__ b_sw,
                            float* __restrict__ pg) {
  int b = blockIdx.x;
  float sum = 0.0f;
  for (int j = threadIdx.x; j < H_; j += 256) sum += h[b * H_ + j] * W_sw[j];
  __shared__ float red[256];
  red[threadIdx.x] = sum;
  __syncthreads();
  for (int off = 128; off > 0; off >>= 1) {
    if (threadIdx.x < off) red[threadIdx.x] += red[threadIdx.x + off];
    __syncthreads();
  }
  if (threadIdx.x == 0) pg[b] = 1.0f / (1.0f + __expf(-(red[0] + b_sw[0])));
}

// Softmax over V on logits and write p_gen*gen into out[b,t,:], zeros in OOV.
__global__ void out_gen_kernel(const float* __restrict__ logits,
                               const float* __restrict__ pg,
                               float* __restrict__ out, int t) {
  int b = blockIdx.x;
  const float* l = logits + (size_t)b * V_;
  const int NT = 256;
  int tid = threadIdx.x;
  __shared__ float red[NT];
  float m = -3.4e38f;
  for (int j = tid; j < V_; j += NT) m = fmaxf(m, l[j]);
  red[tid] = m;
  __syncthreads();
  for (int off = NT / 2; off > 0; off >>= 1) {
    if (tid < off) red[tid] = fmaxf(red[tid], red[tid + off]);
    __syncthreads();
  }
  m = red[0];
  __syncthreads();
  float s = 0.0f;
  for (int j = tid; j < V_; j += NT) s += __expf(l[j] - m);
  red[tid] = s;
  __syncthreads();
  for (int off = NT / 2; off > 0; off >>= 1) {
    if (tid < off) red[tid] += red[tid + off];
    __syncthreads();
  }
  s = red[0];
  float inv = pg[b] / s;
  float* o = out + ((size_t)b * T_ + t) * VT_;
  for (int j = tid; j < V_; j += NT) o[j] = __expf(l[j] - m) * inv;
  for (int j = V_ + tid; j < VT_; j += NT) o[j] = 0.0f;
}

// Scatter copy mass: out[b,t,map[b,s]] += (1-p_gen[b]) * alpha_c[b,s]
__global__ void scatter_copy_kernel(const float* __restrict__ alpha_c,
                                    const float* __restrict__ pg,
                                    const int* __restrict__ map,
                                    float* __restrict__ out, int t) {
  int b = blockIdx.x;
  int s = threadIdx.x;  // 256 == S
  float w = (1.0f - pg[b]) * alpha_c[b * S_ + s];
  int j = map[b * S_ + s];
  atomicAdd(out + ((size_t)b * T_ + t) * VT_ + j, w);
}

// ---------------------------------------------------------------------------
// Host-side orchestration
// ---------------------------------------------------------------------------
static inline void launch_cvt(const float* in, bf16_t* out, int n,
                              hipStream_t stream) {
  cvt_bf16_kernel<<<(n + 255) / 256, 256, 0, stream>>>(in, out, n);
}

// M must be a multiple of 32; each wave computes a 32x16 output block (MT=2),
// reusing every weight fragment twice.
static inline void launch_gemm(const bf16_t* A, const bf16_t* W,
                               const float* bias, float* C, int M, int N,
                               int K, bool accum, hipStream_t stream) {
  dim3 block(128);                       // 4 waves of 32
  dim3 grid((N / 16 + 3) / 4, M / 32);   // waves tile N, blocks tile M (MT=2)
  if (accum) {
    wmma_gemm_kernel<false, true, 2><<<grid, block, 0, stream>>>(A, W, bias, C, M, N, K);
  } else if (bias) {
    wmma_gemm_kernel<true, false, 2><<<grid, block, 0, stream>>>(A, W, bias, C, M, N, K);
  } else {
    wmma_gemm_kernel<false, false, 2><<<grid, block, 0, stream>>>(A, W, bias, C, M, N, K);
  }
}

extern "C" void kernel_launch(void* const* d_in, const int* in_sizes, int n_in,
                              void* d_out, int out_size, void* d_ws,
                              size_t ws_size, hipStream_t stream) {
  (void)in_sizes; (void)n_in; (void)out_size; (void)ws_size;
  const float* questions = (const float*)d_in[0];
  const float* h0        = (const float*)d_in[1];
  const float* c0        = (const float*)d_in[2];
  const float* nodes     = (const float*)d_in[3];
  const unsigned char* mask      = (const unsigned char*)d_in[4];
  const unsigned char* copy_mask = (const unsigned char*)d_in[5];
  const int*   src2trg   = (const int*)d_in[6];
  const float* W_ih = (const float*)d_in[7];
  const float* W_hh = (const float*)d_in[8];
  const float* b_ih = (const float*)d_in[9];
  const float* b_hh = (const float*)d_in[10];
  const float* Wq_a = (const float*)d_in[11];
  const float* Wk_a = (const float*)d_in[12];
  const float* v_a  = (const float*)d_in[13];
  const float* Wq_c = (const float*)d_in[14];
  const float* Wk_c = (const float*)d_in[15];
  const float* v_c  = (const float*)d_in[16];
  const float* W_sw = (const float*)d_in[17];
  const float* b_sw = (const float*)d_in[18];
  const float* W_out = (const float*)d_in[19];
  const float* b_out = (const float*)d_in[20];
  const float* W_prj = (const float*)d_in[21];
  const float* b_prj = (const float*)d_in[22];
  float* out = (float*)d_out;

  // Bump-allocate workspace (256B aligned chunks).
  char* ws = (char*)d_ws;
  size_t off = 0;
  auto alloc = [&](size_t bytes) -> void* {
    void* p = ws + off;
    off = (off + bytes + 255) & ~(size_t)255;
    return p;
  };

  // bf16 weight / activation buffers
  bf16_t* nodes_bf = (bf16_t*)alloc((size_t)B_ * S_ * H_ * 2);
  bf16_t* Wq_a_bf  = (bf16_t*)alloc((size_t)H_ * H_ * 2);
  bf16_t* Wq_c_bf  = (bf16_t*)alloc((size_t)H_ * H_ * 2);
  bf16_t* Wk_a_bf  = (bf16_t*)alloc((size_t)H_ * H_ * 2);
  bf16_t* Wk_c_bf  = (bf16_t*)alloc((size_t)H_ * H_ * 2);
  bf16_t* W_prj_bf = (bf16_t*)alloc((size_t)H_ * (DIN_ + H_) * 2);
  bf16_t* W_ih_bf  = (bf16_t*)alloc((size_t)4 * H_ * H_ * 2);
  bf16_t* W_hh_bf  = (bf16_t*)alloc((size_t)4 * H_ * H_ * 2);
  bf16_t* W_out_bf = (bf16_t*)alloc((size_t)V_ * H_ * 2);
  bf16_t* h_bf     = (bf16_t*)alloc((size_t)B_ * H_ * 2);
  bf16_t* xin_bf   = (bf16_t*)alloc((size_t)B_ * (DIN_ + H_) * 2);
  bf16_t* xi_bf    = (bf16_t*)alloc((size_t)B_ * H_ * 2);
  // f32 buffers
  float* keys_a = (float*)alloc((size_t)B_ * S_ * H_ * 4);
  float* keys_c = (float*)alloc((size_t)B_ * S_ * H_ * 4);
  float* h      = (float*)alloc((size_t)B_ * H_ * 4);
  float* c      = (float*)alloc((size_t)B_ * H_ * 4);
  float* hq_a   = (float*)alloc((size_t)B_ * H_ * 4);
  float* hq_c   = (float*)alloc((size_t)B_ * H_ * 4);
  float* sc_a   = (float*)alloc((size_t)B_ * S_ * 4);
  float* sc_c   = (float*)alloc((size_t)B_ * S_ * 4);
  float* ctx    = (float*)alloc((size_t)B_ * H_ * 4);
  float* xi     = (float*)alloc((size_t)B_ * H_ * 4);
  float* gates  = (float*)alloc((size_t)B_ * 4 * H_ * 4);
  float* logits = (float*)alloc((size_t)B_ * V_ * 4);
  float* pg     = (float*)alloc((size_t)B_ * 4);
  float* b_gates = (float*)alloc((size_t)4 * H_ * 4);

  // ---- setup phase ----
  launch_cvt(nodes, nodes_bf, B_ * S_ * H_, stream);
  launch_cvt(Wq_a, Wq_a_bf, H_ * H_, stream);
  launch_cvt(Wq_c, Wq_c_bf, H_ * H_, stream);
  launch_cvt(Wk_a, Wk_a_bf, H_ * H_, stream);
  launch_cvt(Wk_c, Wk_c_bf, H_ * H_, stream);
  launch_cvt(W_prj, W_prj_bf, H_ * (DIN_ + H_), stream);
  launch_cvt(W_ih, W_ih_bf, 4 * H_ * H_, stream);
  launch_cvt(W_hh, W_hh_bf, 4 * H_ * H_, stream);
  launch_cvt(W_out, W_out_bf, V_ * H_, stream);
  add_bias_kernel<<<(4 * H_ + 255) / 256, 256, 0, stream>>>(b_ih, b_hh,
                                                            b_gates, 4 * H_);
  // Pre-projected attention keys: [B*S, H] = nodes_bf @ Wk^T
  launch_gemm(nodes_bf, Wk_a_bf, nullptr, keys_a, B_ * S_, H_, H_, false, stream);
  launch_gemm(nodes_bf, Wk_c_bf, nullptr, keys_c, B_ * S_, H_, H_, false, stream);
  // Initial state
  hipMemcpyAsync(h, h0, (size_t)B_ * H_ * 4, hipMemcpyDeviceToDevice, stream);
  hipMemcpyAsync(c, c0, (size_t)B_ * H_ * 4, hipMemcpyDeviceToDevice, stream);
  launch_cvt(h0, h_bf, B_ * H_, stream);

  // ---- time loop ----
  for (int t = 0; t < T_; ++t) {
    // Queries from pre-step hidden state
    launch_gemm(h_bf, Wq_a_bf, nullptr, hq_a, B_, H_, H_, false, stream);
    launch_gemm(h_bf, Wq_c_bf, nullptr, hq_c, B_, H_, H_, false, stream);
    // p_gen uses pre-step hidden state (before it is overwritten)
    pgen_kernel<<<B_, 256, 0, stream>>>(h, W_sw, b_sw, pg);
    // Context attention + context vector
    attn_score_kernel<<<B_ * S_, 256, 0, stream>>>(hq_a, keys_a, v_a, mask, sc_a);
    softmax_s_kernel<<<B_, 256, 0, stream>>>(sc_a);
    ctx_kernel<<<B_, 256, 0, stream>>>(sc_a, nodes, ctx);
    // Copy attention (also uses pre-step hidden)
    attn_score_kernel<<<B_ * S_, 256, 0, stream>>>(hq_c, keys_c, v_c, copy_mask, sc_c);
    softmax_s_kernel<<<B_, 256, 0, stream>>>(sc_c);
    // Input projection: xi = [x_t, ctx] @ W_prj^T + b_prj
    concat_kernel<<<(B_ * (DIN_ + H_) + 255) / 256, 256, 0, stream>>>(
        questions, ctx, xin_bf, t);
    launch_gemm(xin_bf, W_prj_bf, b_prj, xi, B_, H_, DIN_ + H_, false, stream);
    launch_cvt(xi, xi_bf, B_ * H_, stream);
    // LSTM gates: xi @ W_ih^T + h @ W_hh^T + (b_ih+b_hh)
    launch_gemm(xi_bf, W_ih_bf, b_gates, gates, B_, 4 * H_, H_, false, stream);
    launch_gemm(h_bf, W_hh_bf, nullptr, gates, B_, 4 * H_, H_, true, stream);
    lstm_cell_kernel<<<(B_ * H_ + 255) / 256, 256, 0, stream>>>(gates, h, c, h_bf);
    // Generation distribution from new hidden state
    launch_gemm(h_bf, W_out_bf, b_out, logits, B_, V_, H_, false, stream);
    out_gen_kernel<<<B_, 256, 0, stream>>>(logits, pg, out, t);
    scatter_copy_kernel<<<B_, S_, 0, stream>>>(sc_c, pg, src2trg, out, t);
  }

  // Final hidden / cell state appended after [B,T,VT] outputs.
  float* out_h = out + (size_t)B_ * T_ * VT_;
  float* out_c = out_h + (size_t)B_ * H_;
  hipMemcpyAsync(out_h, h, (size_t)B_ * H_ * 4, hipMemcpyDeviceToDevice, stream);
  hipMemcpyAsync(out_c, c, (size_t)B_ * H_ * 4, hipMemcpyDeviceToDevice, stream);
}
